// GLMFISelfAttention_22033182228760
// MI455X (gfx1250) — matrix operand order repaired and checked by reading, not compiled, not run on hardware
//
#include <hip/hip_runtime.h>
#include <hip/hip_bf16.h>
#include <math.h>

// ---------------------------------------------------------------------------
// GLM-FI structural self-attention for MI455X (gfx1250, wave32, WMMA bf16,
// Tensor Data Mover for K tiles, global_prefetch for the adjacency stream)
// B=2 S=2048 H=768 N=12 D=64 R=5
// ---------------------------------------------------------------------------

typedef __attribute__((ext_vector_type(16))) __bf16       v16bf;
typedef __attribute__((ext_vector_type(8)))  float        v8f;
typedef __attribute__((ext_vector_type(4)))  unsigned int u32x4;
typedef __attribute__((ext_vector_type(8)))  int          i32x8;
typedef __attribute__((ext_vector_type(4)))  int          i32x4;

#define BC 2
#define SC 2048
#define HC 768
#define NH 12
#define DH 64
#define RC 5
#define MC (BC*SC)   // 4096

#if defined(__has_builtin)
#if __has_builtin(__builtin_amdgcn_tensor_load_to_lds) && \
    __has_builtin(__builtin_amdgcn_s_wait_tensorcnt)
#define HAVE_TDM 1
#endif
#endif

__device__ __forceinline__ __bf16 f2bf(float f) {
    unsigned u = __builtin_bit_cast(unsigned, f);
    unsigned r = u + 0x7FFFu + ((u >> 16) & 1u);   // round-to-nearest-even
    unsigned short h = (unsigned short)(r >> 16);
    return __builtin_bit_cast(__bf16, h);
}

__device__ __forceinline__ v8f wmma_bf16(v16bf a, v16bf b, v8f c) {
    // (neg_a, A, neg_b, B, c_mod, C, reuse_a, reuse_b)
    return __builtin_amdgcn_wmma_f32_16x16x32_bf16(false, a, false, b,
                                                   (short)0, c, false, false);
}

// A fragment (16x32, bf16, row-major source, leading dim ld).
// lane<16 : row=lane,    elems = K[k0+0..15]
// lane>=16: row=lane-16, elems = K[k0+16..31]
__device__ __forceinline__ v16bf load_a_frag(const __bf16* base, int ld, int k0) {
    int lane = threadIdx.x & 31;
    int half = lane >> 4;
    const __bf16* p = base + (size_t)(lane & 15) * ld;
    v16bf a;
#pragma unroll
    for (int i = 0; i < 8; ++i) a[i]     = p[k0 + half * 16 + i];
#pragma unroll
    for (int i = 0; i < 8; ++i) a[8 + i] = p[k0 + half * 16 + 8 + i];
    return a;
}

// B fragment (32x16), matrix stored as [N][K] row-major (X @ W^T with W
// row-major). Contiguous 16-element load per lane.
__device__ __forceinline__ v16bf load_b_frag_nk(const __bf16* base, int ld, int k0) {
    int lane = threadIdx.x & 31;
    int half = lane >> 4;
    const __bf16* p = base + (size_t)(lane & 15) * ld + k0 + half * 16;
    v16bf b;
#pragma unroll
    for (int i = 0; i < 16; ++i) b[i] = p[i];
    return b;
}

// B fragment (32x16), matrix stored as [K][N] row-major.
__device__ __forceinline__ v16bf load_b_frag_kn(const __bf16* base, int ld, int k0) {
    int lane = threadIdx.x & 31;
    int half = lane >> 4;
    const __bf16* p = base + (size_t)(k0 + half * 16) * ld + (lane & 15);
    v16bf b;
#pragma unroll
    for (int i = 0; i < 16; ++i) b[i] = p[(size_t)i * ld];
    return b;
}

// ---------------------------------------------------------------------------
__global__ void cvt_f32_bf16(const float* __restrict__ src,
                             __bf16* __restrict__ dst, int n) {
    int i = blockIdx.x * 256 + threadIdx.x;
    if (i < n) dst[i] = f2bf(src[i]);
}

// ---------------------------------------------------------------------------
// QKV projection: Y[which] = X[4096x768] @ W[which]^T + b[which]  (bf16 out)
// Each wave computes a 16x64 strip: one A-fragment feeds 4 WMMAs per k-step.
// grid: x = (256 m-tiles * 12 n64-tiles)/8 = 384, y = 3 (q,k,v). block = 256.
__global__ void qkv_gemm(const __bf16* __restrict__ X,
                         const __bf16* __restrict__ Wall,
                         const float* __restrict__ bq,
                         const float* __restrict__ bk,
                         const float* __restrict__ bv,
                         __bf16* __restrict__ Yall) {
    const int which = blockIdx.y;
    const __bf16* W = Wall + (size_t)which * HC * HC;
    const float* bias = (which == 0) ? bq : ((which == 1) ? bk : bv);
    __bf16* Y = Yall + (size_t)which * MC * HC;

    const int wid = blockIdx.x * 8 + (threadIdx.x >> 5);
    const int tm   = wid / 12;          // 256 row tiles
    const int tn64 = wid % 12;          // 12 column strips of 64

    const __bf16* Abase = X + (size_t)tm * 16 * HC;

    v8f acc[4] = {v8f{}, v8f{}, v8f{}, v8f{}};
    for (int k0 = 0; k0 < HC; k0 += 32) {
        v16bf aF = load_a_frag(Abase, HC, k0);
#pragma unroll
        for (int c = 0; c < 4; ++c)
            acc[c] = wmma_bf16(aF,
                load_b_frag_nk(W + (size_t)(tn64 * 64 + c * 16) * HC, HC, k0),
                acc[c]);
    }

    const int lane = threadIdx.x & 31, half = lane >> 4, ln = lane & 15;
#pragma unroll
    for (int c = 0; c < 4; ++c) {
        const int col = tn64 * 64 + c * 16 + ln;
        const float bval = bias[col];
#pragma unroll
        for (int v = 0; v < 8; ++v)
            Y[(size_t)(tm * 16 + v + 8 * half) * HC + col] =
                f2bf(acc[c][v] + bval);
    }
}

// ---------------------------------------------------------------------------
// Fused structural flash-attention with in-kernel q@bili_r prologue.
// grid: x = S/128 = 16 i-blocks, y = B*N = 24. block = 256 (8 waves).
__global__ void __launch_bounds__(256)
attn_kernel(const __bf16* __restrict__ Q2,
            const __bf16* __restrict__ K2,
            const __bf16* __restrict__ V2,
            const __bf16* __restrict__ Bili,
            const float* __restrict__ adj,
            const float* __restrict__ mask,
            const float* __restrict__ absb,
            float* __restrict__ out) {
    __shared__ __attribute__((aligned(32))) __bf16 kT[32][DH];   // K tile [j][d]
    __shared__ __attribute__((aligned(32))) __bf16 vTt[DH][32];  // V tile [d][j]
    __shared__ __attribute__((aligned(32))) __bf16 pT[8][16][DH];// per-wave scratch

    const int bn = blockIdx.y;
    const int b = bn / NH, head = bn % NH;
    const int wave = threadIdx.x >> 5;
    const int lane = threadIdx.x & 31, half = lane >> 4, ln = lane & 15;
    const int iw = blockIdx.x * 128 + wave * 16;      // wave's first row

    float ab[RC];
#pragma unroll
    for (int r = 0; r < RC; ++r) ab[r] = absb[r * NH + head];

    // Resident q A-fragments
    const __bf16* qbase = Q2 + ((size_t)(b * SC + iw)) * HC + head * DH;
    v16bf qA0 = load_a_frag(qbase, HC, 0);
    v16bf qA1 = load_a_frag(qbase, HC, 32);

    // ---- prologue: qr[r] = q @ bili[r,head] on the matrix cores, then
    // reshape the C-layout result into A-fragments through per-wave LDS ----
    v16bf rA[RC][2];
#pragma unroll
    for (int r = 0; r < RC; ++r) {
        const __bf16* bm = Bili + ((size_t)(r * NH + head)) * DH * DH;
        v8f acc[4];
#pragma unroll
        for (int c = 0; c < 4; ++c) {
            v8f a = {};
            a = wmma_bf16(qA0, load_b_frag_kn(bm + c * 16, DH, 0),  a);
            a = wmma_bf16(qA1, load_b_frag_kn(bm + c * 16, DH, 32), a);
            acc[c] = a;
        }
        __syncthreads();
#pragma unroll
        for (int c = 0; c < 4; ++c)
#pragma unroll
            for (int v = 0; v < 8; ++v)
                pT[wave][v + 8 * half][c * 16 + ln] = f2bf(acc[c][v]);
        __syncthreads();
        rA[r][0] = load_a_frag(&pT[wave][0][0], DH, 0);
        rA[r][1] = load_a_frag(&pT[wave][0][0], DH, 32);
    }

    v8f ctx[4] = {v8f{}, v8f{}, v8f{}, v8f{}};
    float m_[8], l_[8];
#pragma unroll
    for (int v = 0; v < 8; ++v) { m_[v] = -1.0e30f; l_[v] = 0.0f; }

    for (int j0 = 0; j0 < SC; j0 += 32) {
        __syncthreads();   // previous iteration's LDS reads complete

#ifdef HAVE_TDM
        // K tile via Tensor Data Mover: 32x64 bf16 2-D tile, row stride 768.
        if (wave == 0) {
            unsigned lds_off = (unsigned)(uintptr_t)&kT[0][0];
            unsigned long long ga = (unsigned long long)(uintptr_t)
                (K2 + (size_t)(b * SC + j0) * HC + head * DH);
            u32x4 g0;
            g0[0] = 1u;                               // count=1, user descriptor
            g0[1] = lds_off;                          // lds_addr
            g0[2] = (unsigned)ga;                     // global_addr[31:0]
            g0[3] = (unsigned)((ga >> 32) & 0x1FFFFFFu) | (2u << 30); // type=2
            i32x8 g1;
            g1[0] = (int)(1u << 16);                  // data_size = 2 bytes
            g1[1] = (int)(64u << 16);                 // tensor_dim0 = 64
            g1[2] = (int)(32u << 16);                 // tensor_dim1 = 32
            g1[3] = (int)(64u << 16);                 // tile_dim0 = 64
            g1[4] = 32;                               // tile_dim1 = 32
            g1[5] = HC;                               // tensor_dim0_stride = 768
            g1[6] = 0;
            g1[7] = 0;
            i32x4 z4 = {0, 0, 0, 0};
#if defined(__clang_major__) && __clang_major__ >= 23
            i32x8 z8 = {0, 0, 0, 0, 0, 0, 0, 0};
            __builtin_amdgcn_tensor_load_to_lds(g0, g1, z4, z4, z8, 0);
#else
            __builtin_amdgcn_tensor_load_to_lds(g0, g1, z4, z4, 0);
#endif
        }
        // V tile cooperatively, transposed to [d][j] for contiguous B-frags
        for (int idx = threadIdx.x; idx < 32 * DH; idx += 256) {
            int row = idx >> 6, col = idx & 63;
            vTt[col][row] = V2[(size_t)(b * SC + j0 + row) * HC + head * DH + col];
        }
        if (wave == 0) __builtin_amdgcn_s_wait_tensorcnt(0);
        __syncthreads();
#else
        for (int idx = threadIdx.x; idx < 32 * DH; idx += 256) {
            int row = idx >> 6, col = idx & 63;
            size_t g = (size_t)(b * SC + j0 + row) * HC + head * DH + col;
            kT[row][col]  = K2[g];
            vTt[col][row] = V2[g];
        }
        __syncthreads();
#endif

        // Prefetch next j-tile of the dominant adjacency stream (HBM-bound)
        if (j0 + 32 < SC) {
#pragma unroll
            for (int r = 0; r < RC; ++r)
                __builtin_prefetch(adj
                    + (((size_t)r * BC + b) * SC + (iw + ln)) * SC
                    + j0 + 32 + half * 16, 0, 3);
        }

        // K^T B-fragments shared by all 6 score GEMMs (aligned 32B LDS loads)
        v16bf kB[2][2];
#pragma unroll
        for (int c = 0; c < 2; ++c)
#pragma unroll
            for (int kk = 0; kk < 2; ++kk)
                kB[c][kk] =
                    *(const v16bf*)&kT[c * 16 + ln][kk * 32 + half * 16];

        v8f s[2];
#pragma unroll
        for (int c = 0; c < 2; ++c) {
            v8f acc = {};
            acc = wmma_bf16(qA0, kB[c][0], acc);
            acc = wmma_bf16(qA1, kB[c][1], acc);
#pragma unroll
            for (int r = 0; r < RC; ++r) {
                v8f bc = {};
                bc = wmma_bf16(rA[r][0], kB[c][0], bc);
                bc = wmma_bf16(rA[r][1], kB[c][1], bc);
                const float* ap = adj
                    + (((size_t)r * BC + b) * SC + (iw + 8 * half)) * SC
                    + j0 + c * 16 + ln;
#pragma unroll
                for (int v = 0; v < 8; ++v)
                    acc[v] += ap[(size_t)v * SC] * (bc[v] + ab[r]);
            }
            const float mk = mask[b * SC + j0 + c * 16 + ln];
#pragma unroll
            for (int v = 0; v < 8; ++v) s[c][v] = acc[v] * 0.125f + mk;
        }

        // ---- online softmax (rows span the 16 lanes of each half) ----
        float rm[8];
#pragma unroll
        for (int v = 0; v < 8; ++v) {
            rm[v] = fmaxf(s[0][v], s[1][v]);
            rm[v] = fmaxf(rm[v], __shfl_xor(rm[v], 8, 32));
            rm[v] = fmaxf(rm[v], __shfl_xor(rm[v], 4, 32));
            rm[v] = fmaxf(rm[v], __shfl_xor(rm[v], 2, 32));
            rm[v] = fmaxf(rm[v], __shfl_xor(rm[v], 1, 32));
        }
        float corr[8];
#pragma unroll
        for (int v = 0; v < 8; ++v) {
            float mn = fmaxf(m_[v], rm[v]);
            corr[v] = __expf(m_[v] - mn);
            m_[v] = mn;
            float p0 = __expf(s[0][v] - mn);
            float p1 = __expf(s[1][v] - mn);
            s[0][v] = p0; s[1][v] = p1;
            float rs = p0 + p1;
            rs += __shfl_xor(rs, 8, 32);
            rs += __shfl_xor(rs, 4, 32);
            rs += __shfl_xor(rs, 2, 32);
            rs += __shfl_xor(rs, 1, 32);
            l_[v] = l_[v] * corr[v] + rs;
        }
#pragma unroll
        for (int c = 0; c < 4; ++c)
#pragma unroll
            for (int v = 0; v < 8; ++v) ctx[c][v] *= corr[v];

        // ---- P through LDS: C-layout -> A-layout reshape ----
#pragma unroll
        for (int c = 0; c < 2; ++c)
#pragma unroll
            for (int v = 0; v < 8; ++v)
                pT[wave][v + 8 * half][c * 16 + ln] = f2bf(s[c][v]);
        __syncthreads();

        v16bf pA = load_a_frag(&pT[wave][0][0], DH, 0);
        // P[16x32] @ V[32x64] with contiguous V B-fragments
#pragma unroll
        for (int nb = 0; nb < 4; ++nb) {
            v16bf vB = *(const v16bf*)&vTt[nb * 16 + ln][half * 16];
            ctx[nb] = wmma_bf16(pA, vB, ctx[nb]);
        }
    }

    // ---- epilogue: normalize and store [B,S,H] fp32 ----
    float invl[8];
#pragma unroll
    for (int v = 0; v < 8; ++v) invl[v] = 1.0f / l_[v];
#pragma unroll
    for (int c = 0; c < 4; ++c)
#pragma unroll
        for (int v = 0; v < 8; ++v)
            out[((size_t)(b * SC + iw + v + 8 * half)) * HC
                + head * DH + c * 16 + ln] = ctx[c][v] * invl[v];
}

// ---------------------------------------------------------------------------
extern "C" void kernel_launch(void* const* d_in, const int* in_sizes, int n_in,
                              void* d_out, int out_size, void* d_ws, size_t ws_size,
                              hipStream_t stream) {
    (void)in_sizes; (void)n_in; (void)out_size; (void)ws_size;
    const float* hidden = (const float*)d_in[0];
    const float* mask   = (const float*)d_in[1];
    const float* adj    = (const float*)d_in[2];
    const float* Wq     = (const float*)d_in[3];
    const float* bq     = (const float*)d_in[4];
    const float* Wk     = (const float*)d_in[5];
    const float* bk     = (const float*)d_in[6];
    const float* Wv     = (const float*)d_in[7];
    const float* bv     = (const float*)d_in[8];
    const float* bili   = (const float*)d_in[9];
    const float* absb   = (const float*)d_in[10];
    float* out = (float*)d_out;

    // Workspace layout (bf16 staging), ~29 MB total
    char* w = (char*)d_ws;
    __bf16* hb    = (__bf16*)w;  w += (size_t)MC * HC * 2;        // hidden bf16
    __bf16* Wb    = (__bf16*)w;  w += (size_t)3 * HC * HC * 2;    // Wq|Wk|Wv bf16
    __bf16* bilib = (__bf16*)w;  w += (size_t)RC * NH * DH * DH * 2;
    __bf16* qkv   = (__bf16*)w;  w += (size_t)3 * MC * HC * 2;    // q2|k2|v2 bf16

    auto cvt = [&](const float* s, __bf16* d, int n) {
        cvt_f32_bf16<<<(n + 255) / 256, 256, 0, stream>>>(s, d, n);
    };
    cvt(hidden, hb, MC * HC);
    cvt(Wq, Wb,                   HC * HC);
    cvt(Wk, Wb + (size_t)HC * HC, HC * HC);
    cvt(Wv, Wb + (size_t)2 * HC * HC, HC * HC);
    cvt(bili, bilib, RC * NH * DH * DH);

    // QKV: 256 m-tiles * 12 n64-strips = 3072 waves / 8 = 384 blocks, y=3
    qkv_gemm<<<dim3(384, 3), 256, 0, stream>>>(hb, Wb, bq, bk, bv, qkv);

    // Fused attention: 16 i-blocks x 24 (b,head)
    attn_kernel<<<dim3(SC / 128, BC * NH), 256, 0, stream>>>(
        qkv,                       // Q
        qkv + (size_t)MC * HC,     // K
        qkv + (size_t)2 * MC * HC, // V
        bilib, adj, mask, absb, out);
}
